// ModulatedDeformConv3d_84774064489048
// MI455X (gfx1250) — matrix operand order repaired and checked by reading, not compiled, not run on hardware
//
#include <hip/hip_runtime.h>
#include <hip/hip_bf16.h>

typedef __bf16 bf16_t;
typedef __attribute__((ext_vector_type(16))) __bf16 v16bf;
typedef __attribute__((ext_vector_type(8)))  __bf16 v8bf;
typedef __attribute__((ext_vector_type(8)))  float  v8f;

#define PTOT  32768         // Ho*Wo*Lo = 32^3
#define KTOT  1728          // C * 27
#define CIN   64
#define COUT  64
#define NPTS  32            // output points per workgroup (two WMMA N-tiles)
#define NPAIR (27 * NPTS)   // 864 (tap, point) pairs
#define NTHREADS 128        // 4 waves; wave w owns Cout rows [16w, 16w+16)
#define WELEM (COUT * KTOT) // 110592 weight elements

// ---- prep: convert weight matrix f32 -> bf16 once into workspace ----------
__global__ __launch_bounds__(256)
void wconv_bf16_kernel(const float* __restrict__ wgt, bf16_t* __restrict__ wbf) {
  const int i = (blockIdx.x * 256 + threadIdx.x) * 4;
  const float4 w = *(const float4*)(wgt + i);
  v8bf* dst = (v8bf*)wbf;   // write 4 bf16 = 8 bytes via half-vector
  bf16_t o[4] = {(bf16_t)w.x, (bf16_t)w.y, (bf16_t)w.z, (bf16_t)w.w};
  *(uint2*)(wbf + i) = *(const uint2*)o;
  (void)dst;
}

// ---- fused deformable-sampling + bf16 WMMA GEMM ---------------------------
__global__ __launch_bounds__(NTHREADS)
void dcn3d_wmma_kernel(const float* __restrict__ x,
                       const float* __restrict__ off,
                       const float* __restrict__ msk,
                       const bf16_t* __restrict__ wbf,
                       float* __restrict__ out) {
  // bf16 column tile, point-major so B-fragments are contiguous K runs.
  __shared__ __align__(32) bf16_t s_col[NPTS * KTOT];  // 110592 bytes

  const int tid   = threadIdx.x;
  const int pbase = blockIdx.x * NPTS;

  // ---------------- Phase 1: deformable trilinear sampling -> LDS (bf16) ---
  for (int t = tid; t < NPAIR; t += NTHREADS) {
    const int k = t >> 5;              // tap 0..26
    const int n = t & 31;              // point within tile
    const int p = pbase + n;
    const int lo = p & 31, wo = (p >> 5) & 31, ho = p >> 10;

    // offset layout: [k*3 + {h,w,l}][p]; kernel tap decomposed (ki,kj,kk)
    const float chh = off[(k * 3 + 0) * PTOT + p] + (float)(ho - 1 + (k / 9));
    const float cww = off[(k * 3 + 1) * PTOT + p] + (float)(wo - 1 + ((k / 3) % 3));
    const float cll = off[(k * 3 + 2) * PTOT + p] + (float)(lo - 1 + (k % 3));
    const float m   = msk[k * PTOT + p];

    const float h0 = floorf(chh), w0 = floorf(cww), l0 = floorf(cll);
    const float fh = chh - h0, fw = cww - w0, fl = cll - l0;
    const int h0i = (int)h0, w0i = (int)w0, l0i = (int)l0;

    int   cidx[8];
    float cwgt[8];
#pragma unroll
    for (int j = 0; j < 8; ++j) {
      const int dh = (j >> 2) & 1, dw = (j >> 1) & 1, dl = j & 1;
      const int ih = h0i + dh, iw = w0i + dw, il = l0i + dl;
      const float wv = (dh ? fh : 1.f - fh) * (dw ? fw : 1.f - fw) *
                       (dl ? fl : 1.f - fl) * m;
      const bool valid = ((unsigned)ih < 32u) & ((unsigned)iw < 32u) &
                         ((unsigned)il < 32u);
      const int ihc = min(max(ih, 0), 31);
      const int iwc = min(max(iw, 0), 31);
      const int ilc = min(max(il, 0), 31);
      cidx[j] = (ihc * 32 + iwc) * 32 + ilc;
      cwgt[j] = valid ? wv : 0.f;
    }

    bf16_t* colw = &s_col[n * KTOT + k];   // stride 27 over channels
#pragma unroll 2
    for (int c = 0; c < CIN; ++c) {
      const float* xc = x + c * PTOT;      // L2-resident (x = 8 MB << 192 MB)
      float acc = 0.f;
#pragma unroll
      for (int j = 0; j < 8; ++j)
        acc += xc[cidx[j]] * cwgt[j];
      colw[c * 27] = (bf16_t)acc;
    }
  }
  __syncthreads();

  // ---------------- Phase 2: GEMM via V_WMMA_F32_16X16X32_BF16 -------------
  const int lane = tid & 31;
  const int wave = tid >> 5;                       // Cout tile 0..3
  const int hi   = (lane & 16) ? 1 : 0;            // upper half-wave

  // A (16x32 bf16): lanes 0-15 hold K 0..15; lanes 16-31 hold
  // K 16..31; both halves cover M = lane%16.  Loaded as bf16.
  const bf16_t* wr = wbf + (wave * 16 + (lane & 15)) * KTOT + (hi ? 16 : 0);
  // B (32x16 bf16): lanes 0-15 -> K 0..15, lanes 16-31 -> K 16..31,
  // N = lane%16.  Two N-tiles share the same A fragment.
  const bf16_t* br0 = &s_col[(lane & 15) * KTOT + (hi ? 16 : 0)];
  const bf16_t* br1 = br0 + 16 * KTOT;

  v8f acc0 = {};
  v8f acc1 = {};
  for (int kk = 0; kk < KTOT; kk += 32) {
    union { v16bf v; v8bf h[2]; } au;
    au.h[0] = *(const v8bf*)(wr + kk);         // K run 1 (16 B)
    au.h[1] = *(const v8bf*)(wr + kk + 8);     // K run 2 (16 B)
    if (kk + 32 < KTOT)
      __builtin_prefetch((const void*)(wr + kk + 32), 0, 0);

    const v16bf b0 = *(const v16bf*)(br0 + kk);   // 2x ds_load_b128
    const v16bf b1 = *(const v16bf*)(br1 + kk);   // 2x ds_load_b128

    acc0 = __builtin_amdgcn_wmma_f32_16x16x32_bf16(
        false, au.v, false, b0, (short)0, acc0, false, false);
    acc1 = __builtin_amdgcn_wmma_f32_16x16x32_bf16(
        false, au.v, false, b1, (short)0, acc1, false, false);
  }

  // C/D layout: VGPR r, lane L -> M = r + 8*(L>=16), N = L%16
  const int p0   = pbase + (lane & 15);
  const int orow = wave * 16 + (hi ? 8 : 0);
#pragma unroll
  for (int r = 0; r < 8; ++r) {
    out[(orow + r) * PTOT + p0]      = acc0[r];
    out[(orow + r) * PTOT + p0 + 16] = acc1[r];
  }
}

extern "C" void kernel_launch(void* const* d_in, const int* in_sizes, int n_in,
                              void* d_out, int out_size, void* d_ws, size_t ws_size,
                              hipStream_t stream) {
  const float* x   = (const float*)d_in[0];   // [1,64,32,32,32]
  const float* off = (const float*)d_in[1];   // [1,81,32,32,32]
  const float* msk = (const float*)d_in[2];   // [1,27,32,32,32]
  const float* wgt = (const float*)d_in[3];   // [64,64,3,3,3] -> [64,1728]
  float* out = (float*)d_out;                 // [1,64,32,32,32]
  bf16_t* wbf = (bf16_t*)d_ws;                // 221184 B bf16 weight copy

  // 1) convert weights to bf16 once (110592 elems, 4 per thread)
  wconv_bf16_kernel<<<WELEM / (256 * 4), 256, 0, stream>>>(wgt, wbf);
  // 2) fused sampling + WMMA GEMM
  dcn3d_wmma_kernel<<<PTOT / NPTS, NTHREADS, 0, stream>>>(x, off, msk, wbf, out);
}
